// MA_69827578298789
// MI455X (gfx1250) — matrix operand order repaired and checked by reading, not compiled, not run on hardware
//
#include <hip/hip_runtime.h>
#include <cfloat>
#include <math.h>

typedef _Float16 v16h __attribute__((ext_vector_type(16)));
typedef _Float16 v8h  __attribute__((ext_vector_type(8)));
typedef _Float16 v4h  __attribute__((ext_vector_type(4)));
typedef float    v8f  __attribute__((ext_vector_type(8)));

#define D_DIM 256
#define B_Q   256
#define K_NN  64
#define CHUNK_COLS 2048
#define SUB   128
#define NSUB  (CHUNK_COLS / SUB)
#define SMEM_BYTES (SUB * D_DIM * 2 + B_Q * SUB * 4)  // 64KB f16 keys + 128KB f32 stage

// ---------------------------------------------------------------------------
// wave32 bitonic top-64 helpers: 64 elements = 2 regs/lane, element e = r*32+lane
// ---------------------------------------------------------------------------
__device__ __forceinline__ void ce_pair(float& v, int& i, int j, bool down, unsigned lane) {
  float pv = __shfl_xor(v, j, 32);
  int   pi = __shfl_xor(i, j, 32);
  bool low = ((lane & (unsigned)j) == 0u);
  bool takeMax = (down == low);
  bool sw = takeMax ? (pv > v) : (pv < v);
  v = sw ? pv : v;
  i = sw ? pi : i;
}

__device__ __forceinline__ void sort64_desc(float v[2], int idx[2], unsigned lane) {
#pragma unroll
  for (int k = 2; k <= 16; k <<= 1) {
#pragma unroll
    for (int j = k >> 1; j >= 1; j >>= 1) {
      bool down = ((lane & (unsigned)k) == 0u);
      ce_pair(v[0], idx[0], j, down, lane);
      ce_pair(v[1], idx[1], j, down, lane);
    }
  }
#pragma unroll
  for (int j = 16; j >= 1; j >>= 1) {   // k = 32: r=0 desc, r=1 asc
    ce_pair(v[0], idx[0], j, true, lane);
    ce_pair(v[1], idx[1], j, false, lane);
  }
  { // k = 64, j = 32: in-lane r swap, max to r=0
    bool sw = (v[1] > v[0]);
    float tv = v[0]; int ti = idx[0];
    v[0] = sw ? v[1] : v[0]; idx[0] = sw ? idx[1] : idx[0];
    v[1] = sw ? tv : v[1];   idx[1] = sw ? ti : idx[1];
  }
#pragma unroll
  for (int j = 16; j >= 1; j >>= 1) {
    ce_pair(v[0], idx[0], j, true, lane);
    ce_pair(v[1], idx[1], j, true, lane);
  }
}

// R desc-sorted, S desc-sorted -> R = top-64 of union (desc-sorted)
__device__ __forceinline__ void merge64_desc(float rv[2], int ri[2],
                                             float sv[2], int si[2], unsigned lane) {
  float s0 = __shfl_xor(sv[1], 31, 32); int j0 = __shfl_xor(si[1], 31, 32);
  float s1 = __shfl_xor(sv[0], 31, 32); int j1 = __shfl_xor(si[0], 31, 32);
  if (s0 > rv[0]) { rv[0] = s0; ri[0] = j0; }
  if (s1 > rv[1]) { rv[1] = s1; ri[1] = j1; }
  { // bitonic clean: j=32 (r swap) then lane stages, all descending
    bool sw = (rv[1] > rv[0]);
    float tv = rv[0]; int ti = ri[0];
    rv[0] = sw ? rv[1] : rv[0]; ri[0] = sw ? ri[1] : ri[0];
    rv[1] = sw ? tv : rv[1];    ri[1] = sw ? ti : ri[1];
  }
#pragma unroll
  for (int j = 16; j >= 1; j >>= 1) {
    ce_pair(rv[0], ri[0], j, true, lane);
    ce_pair(rv[1], ri[1], j, true, lane);
  }
}

// ---------------------------------------------------------------------------
// K1: qe = relu(query); qn_f16 = normalize(qe) as f16
// ---------------------------------------------------------------------------
__global__ void prep_query_kernel(const float* __restrict__ query,
                                  float* __restrict__ qe, _Float16* __restrict__ qn) {
  int q = blockIdx.x, d = threadIdx.x;
  float v = fmaxf(query[q * D_DIM + d], 0.0f);
  qe[q * D_DIM + d] = v;
  __shared__ float red[D_DIM];
  red[d] = v * v;
  __syncthreads();
  for (int s = 128; s > 0; s >>= 1) {
    if (d < s) red[d] += red[d + s];
    __syncthreads();
  }
  float rs = 1.0f / fmaxf(sqrtf(red[0]), 1e-8f);
  qn[q * D_DIM + d] = (_Float16)(v * rs);
}

// ---------------------------------------------------------------------------
// K1b: qt = relu(qe @ W_dense^T + b_dense)   (tiny: 33 MFLOP)
// ---------------------------------------------------------------------------
__global__ void dense_qt_kernel(const float* __restrict__ qe, const float* __restrict__ W,
                                const float* __restrict__ b, float* __restrict__ qt) {
  int q = blockIdx.x, c = threadIdx.x;
  __shared__ float sq[D_DIM];
  sq[c] = qe[q * D_DIM + c];
  __syncthreads();
  float acc = b[c];
  const float* wr = W + c * D_DIM;
#pragma unroll 8
  for (int d = 0; d < D_DIM; ++d) acc = fmaf(sq[d], wr[d], acc);
  qt[q * D_DIM + c] = fmaxf(acc, 0.0f);
}

// ---------------------------------------------------------------------------
// K2: fused normalize(memory_keys) -> f16 LDS -> WMMA sims -> per-chunk top-64
// ---------------------------------------------------------------------------
__global__ void __launch_bounds__(256)
sims_topk_kernel(const float* __restrict__ mk, const _Float16* __restrict__ qn,
                 float* __restrict__ candV, int* __restrict__ candI, long N) {
  extern __shared__ char smem[];
  _Float16* lds_mn = (_Float16*)smem;                       // [SUB][D] f16
  float*    lds_stg = (float*)(smem + SUB * D_DIM * 2);     // [B_Q][SUB] f32

  const int tid = threadIdx.x;
  const unsigned lane = (unsigned)(tid & 31);
  const int wave = tid >> 5;              // 0..7
  const int qbase = wave * 16;
  const long colchunk = (long)blockIdx.x * CHUNK_COLS;

  // A fragments for this wave's 16 query rows (persist across whole block)
  v16h af[8];
  {
    int m = qbase + (int)(lane & 15u);
    int koff = (lane >= 16u) ? 16 : 0;
    const _Float16* qrow = qn + m * D_DIM;
#pragma unroll
    for (int kc = 0; kc < 8; ++kc) {
      const _Float16* p = qrow + kc * 32 + koff;
      v8h lo = *(const v8h*)(p);
      v8h hi = *(const v8h*)(p + 8);
      v16h a;
#pragma unroll
      for (int h = 0; h < 8; ++h) { a[h] = lo[h]; a[h + 8] = hi[h]; }
      af[kc] = a;
    }
  }

  // running sorted top-64 per owned query
  float rv[16][2]; int ri[16][2]; float thr[16];
#pragma unroll
  for (int q = 0; q < 16; ++q) {
    rv[q][0] = rv[q][1] = -FLT_MAX;
    ri[q][0] = ri[q][1] = 0;
    thr[q] = -FLT_MAX;
  }

  for (int sc = 0; sc < NSUB; ++sc) {
    const long colbase = colchunk + (long)sc * SUB;
    __syncthreads();  // protect lds_mn against overwrite while still being read

    // ---- phase A: normalize 128 key rows -> f16 in LDS (2 threads / row) ----
    {
      int row = tid >> 1, half = tid & 1;
      long jrow = colbase + row;
      _Float16* dst = lds_mn + row * D_DIM + half * 128;
      if (jrow < N) {
        const float4* src = (const float4*)(mk + jrow * D_DIM + half * 128);
        float4 buf[32];
        float ss = 0.0f;
#pragma unroll
        for (int i = 0; i < 32; ++i) {
          float4 v = src[i];
          buf[i] = v;
          ss += v.x * v.x + v.y * v.y + v.z * v.z + v.w * v.w;
        }
        ss += __shfl_xor(ss, 1, 32);  // combine the two half-row partials
        float rs = 1.0f / fmaxf(sqrtf(ss), 1e-8f);
#pragma unroll
        for (int i = 0; i < 32; ++i) {
          float4 v = buf[i];
          v4h h;
          h[0] = (_Float16)(v.x * rs); h[1] = (_Float16)(v.y * rs);
          h[2] = (_Float16)(v.z * rs); h[3] = (_Float16)(v.w * rs);
          *(v4h*)(dst + i * 4) = h;
        }
      } else {
        v4h z = {};
#pragma unroll
        for (int i = 0; i < 32; ++i) *(v4h*)(dst + i * 4) = z;
      }
    }
    __syncthreads();

    // ---- phase B: WMMA 16 queries x 128 cols, stage fp32 tile to LDS ----
    float* stg = lds_stg + qbase * SUB;  // wave-private [16][SUB]
#pragma unroll
    for (int ct = 0; ct < 8; ++ct) {
      v8f acc = {};
      int n = ct * 16 + (int)(lane & 15u);
      int kboff = (lane >= 16u) ? 16 : 0;
      const _Float16* bp = lds_mn + n * D_DIM + kboff;
#pragma unroll
      for (int kc = 0; kc < 8; ++kc) {
        v16h bfrag = *(const v16h*)(bp + kc * 32);
        acc = __builtin_amdgcn_wmma_f32_16x16x32_f16(
            false, af[kc], false, bfrag, (short)0, acc, false, false);
      }
      int chalf = (lane >= 16u) ? 8 : 0;
      int c = ct * 16 + (int)(lane & 15u);
#pragma unroll
      for (int r = 0; r < 8; ++r) stg[(r + chalf) * SUB + c] = acc[r];
    }
    asm volatile("s_wait_dscnt 0" ::: "memory");  // wave-local LDS stage visible

    // ---- phase C: fold staged sims into running top-64 (threshold-gated) ----
#pragma unroll
    for (int q = 0; q < 16; ++q) {
      const float* srow = stg + q * SUB;
#pragma unroll
      for (int b2 = 0; b2 < 2; ++b2) {
        int e0 = b2 * 64 + (int)lane;
        int e1 = e0 + 32;
        long g0 = colbase + e0, g1 = colbase + e1;
        float v0 = (g0 < N) ? srow[e0] : -FLT_MAX;
        float v1 = (g1 < N) ? srow[e1] : -FLT_MAX;
        float mx = fmaxf(v0, v1);
#pragma unroll
        for (int s = 16; s >= 1; s >>= 1) mx = fmaxf(mx, __shfl_xor(mx, s, 32));
        if (mx > thr[q]) {  // wave-uniform
          float sv[2] = { v0, v1 };
          int   si[2] = { (int)g0, (int)g1 };
          sort64_desc(sv, si, lane);
          merge64_desc(rv[q], ri[q], sv, si, lane);
          thr[q] = __shfl(rv[q][1], 31, 32);
        }
      }
    }
  }

  // write per-chunk sorted candidates
  {
    long base = ((long)blockIdx.x * B_Q + qbase) * K_NN;
#pragma unroll
    for (int q = 0; q < 16; ++q) {
      long o = base + (long)q * K_NN;
      candV[o + lane]      = rv[q][0];
      candV[o + 32 + lane] = rv[q][1];
      candI[o + lane]      = ri[q][0];
      candI[o + 32 + lane] = ri[q][1];
    }
  }
}

// ---------------------------------------------------------------------------
// K3: merge per-chunk sorted top-64 lists -> final top-64 indices (1 wave/query)
// ---------------------------------------------------------------------------
__global__ void topk_merge_kernel(const float* __restrict__ candV,
                                  const int* __restrict__ candI,
                                  int* __restrict__ topI, int nchunk) {
  int q = blockIdx.x;
  unsigned lane = (unsigned)threadIdx.x;
  float rv[2] = { -FLT_MAX, -FLT_MAX };
  int   ri[2] = { 0, 0 };
  for (int c = 0; c < nchunk; ++c) {
    long o = ((long)c * B_Q + q) * K_NN;
    float sv[2] = { candV[o + lane], candV[o + 32 + lane] };
    int   si[2] = { candI[o + lane], candI[o + 32 + lane] };
    merge64_desc(rv, ri, sv, si, lane);  // inputs already desc-sorted
  }
  topI[q * K_NN + lane]      = ri[0];
  topI[q * K_NN + 32 + lane] = ri[1];
}

// ---------------------------------------------------------------------------
// K4: fused attention + residual LayerNorm + classifier head (1 block/query)
// ---------------------------------------------------------------------------
__global__ void attn_head_kernel(const float* __restrict__ qe, const float* __restrict__ qt,
                                 const float* __restrict__ mk, const int* __restrict__ topI,
                                 const float* __restrict__ gamma, const float* __restrict__ beta,
                                 const float* __restrict__ Wc, const float* __restrict__ bc,
                                 float* __restrict__ out, int C) {
  int q = blockIdx.x, t = threadIdx.x;
  __shared__ int   sIdx[K_NN];
  __shared__ float sS[K_NN];
  __shared__ float sRed[D_DIM];
  __shared__ float sM[2 * D_DIM];
  __shared__ float sStat[2];

  if (t < K_NN) sIdx[t] = topI[q * K_NN + t];
  __syncthreads();

  // scores[j] = qt[q] . mk[idx_j]  (4 threads per j)
  {
    int j = t >> 2, part = t & 3;
    const float* row = mk + (long)sIdx[j] * D_DIM;
    const float* qtq = qt + q * D_DIM;
    float p = 0.0f;
    int d0 = part * 64;
#pragma unroll 8
    for (int d = 0; d < 64; ++d) p = fmaf(qtq[d0 + d], row[d0 + d], p);
    p += __shfl_xor(p, 1, 32);
    p += __shfl_xor(p, 2, 32);
    if (part == 0) sS[j] = p;
  }
  __syncthreads();

  if (t == 0) {
    float m = -FLT_MAX;
    for (int j = 0; j < K_NN; ++j) m = fmaxf(m, sS[j]);
    sStat[0] = m;
  }
  __syncthreads();
  if (t < K_NN) sS[t] = expf(sS[t] - sStat[0]);
  __syncthreads();
  if (t == 0) {
    float s = 0.0f;
    for (int j = 0; j < K_NN; ++j) s += sS[j];
    sStat[1] = s;
  }
  __syncthreads();

  // attended[d] + residual
  float a = 0.0f;
#pragma unroll 8
  for (int j = 0; j < K_NN; ++j) a = fmaf(sS[j], mk[(long)sIdx[j] * D_DIM + t], a);
  float x = a / sStat[1] + qe[q * D_DIM + t];

  // LayerNorm over D
  sRed[t] = x;
  __syncthreads();
  for (int s = 128; s > 0; s >>= 1) { if (t < s) sRed[t] += sRed[t + s]; __syncthreads(); }
  float mu = sRed[0] * (1.0f / D_DIM);
  __syncthreads();
  float dx = x - mu;
  sRed[t] = dx * dx;
  __syncthreads();
  for (int s = 128; s > 0; s >>= 1) { if (t < s) sRed[t] += sRed[t + s]; __syncthreads(); }
  float var = sRed[0] * (1.0f / D_DIM);
  float y = dx * rsqrtf(var + 1e-5f) * gamma[t] + beta[t];

  sM[t] = qe[q * D_DIM + t];
  sM[D_DIM + t] = y;
  __syncthreads();

  if (t < C) {
    float acc = bc[t];
    const float* wr = Wc + t * (2 * D_DIM);
#pragma unroll 8
    for (int i = 0; i < 2 * D_DIM; ++i) acc = fmaf(sM[i], wr[i], acc);
    out[q * C + t] = acc;
  }
}

// ---------------------------------------------------------------------------
extern "C" void kernel_launch(void* const* d_in, const int* in_sizes, int n_in,
                              void* d_out, int out_size, void* d_ws, size_t ws_size,
                              hipStream_t stream) {
  const float* query = (const float*)d_in[0];
  const float* mk    = (const float*)d_in[1];
  const float* Wd    = (const float*)d_in[2];
  const float* bd    = (const float*)d_in[3];
  const float* gamma = (const float*)d_in[4];
  const float* beta  = (const float*)d_in[5];
  const float* Wc    = (const float*)d_in[6];
  const float* bc    = (const float*)d_in[7];

  const int B = in_sizes[0] / D_DIM;          // 256
  const long N = (long)in_sizes[1] / D_DIM;   // 500000
  const int C = out_size / B;                 // 50
  const int nchunk = (int)((N + CHUNK_COLS - 1) / CHUNK_COLS);

  // workspace carving
  char* w = (char*)d_ws;
  float*    qe = (float*)w;      w += (size_t)B * D_DIM * sizeof(float);
  float*    qt = (float*)w;      w += (size_t)B * D_DIM * sizeof(float);
  _Float16* qn = (_Float16*)w;   w += (size_t)B * D_DIM * sizeof(_Float16);
  float* candV = (float*)w;      w += (size_t)nchunk * B * K_NN * sizeof(float);
  int*   candI = (int*)w;        w += (size_t)nchunk * B * K_NN * sizeof(int);
  int*   topI  = (int*)w;        w += (size_t)B * K_NN * sizeof(int);

  (void)hipFuncSetAttribute(reinterpret_cast<const void*>(&sims_topk_kernel),
                            hipFuncAttributeMaxDynamicSharedMemorySize, SMEM_BYTES);

  prep_query_kernel<<<B, D_DIM, 0, stream>>>(query, qe, qn);
  dense_qt_kernel<<<B, D_DIM, 0, stream>>>(qe, Wd, bd, qt);
  sims_topk_kernel<<<nchunk, 256, SMEM_BYTES, stream>>>(mk, qn, candV, candI, N);
  topk_merge_kernel<<<B, 32, 0, stream>>>(candV, candI, topI, nchunk);
  attn_head_kernel<<<B, D_DIM, 0, stream>>>(qe, qt, mk, topI, gamma, beta, Wc, bc,
                                            (float*)d_out, C);
}
